// TFtoTGShortcut_76828374991775
// MI455X (gfx1250) — compile-verified, matrix operand order checked by reading, not run on hardware
//
#include <hip/hip_runtime.h>

// ---------------------------------------------------------------------------
// Problem constants (from reference): G=20000, T=1536, D=512, B=256, TOPK=32
// d_out layout: [B*G] tf_scalar (scaled) ++ [G*T] attn
// ---------------------------------------------------------------------------

typedef __attribute__((ext_vector_type(16))) __bf16        v16bf;
typedef __attribute__((ext_vector_type(8)))  float         v8f;
typedef __attribute__((ext_vector_type(4)))  unsigned int  u32x4;
typedef __attribute__((ext_vector_type(4)))  float         f32x4;

#define NEG_INF (-__builtin_inff())

constexpr int   kG = 20000, kT = 1536, kD = 512, kB = 256, kTopK = 32;
constexpr int   kMT = 16;                         // G-rows per block
constexpr int   kWaves = 8;                       // 256 threads / wave32
constexpr int   kTilesN = kT / 16;                // 96 N-tiles
constexpr int   kTilesPerWave = kTilesN / kWaves; // 12
constexpr int   kSimStride = kT + 4;              // padded f32 row stride in LDS
constexpr float kRsqrtD = 0.04419417382415922f;   // 1/sqrt(512)

union Frag { u32x4 q[2]; v16bf v; };

// ---- Prep: f32 -> bf16 (round-to-nearest-even) ----------------------------
__global__ void cvt_f32_to_bf16(const float* __restrict__ src,
                                unsigned short* __restrict__ dst, int n) {
  int i = blockIdx.x * blockDim.x + threadIdx.x;
  if (i < n) {
    unsigned int u = __float_as_uint(src[i]);
    unsigned int r = u + 0x7FFFu + ((u >> 16) & 1u);
    dst[i] = (unsigned short)(r >> 16);
  }
}

// ---- Prep: tf_expr [B,T] -> tfT [T,B] -------------------------------------
__global__ void transpose_bt(const float* __restrict__ src,
                             float* __restrict__ dst) {
  int i = blockIdx.x * blockDim.x + threadIdx.x;
  if (i < kB * kT) {
    int b = i / kT, t = i - b * kT;
    dst[(size_t)t * kB + b] = src[i];
  }
}

// ---------------------------------------------------------------------------
// Fused: bf16 WMMA GEMM (sim) + mask + softmax/top-32 + attn store + sparse
// combine into tf_scalar. One block = 16 rows of G.
// ---------------------------------------------------------------------------
__global__ __launch_bounds__(256) void fused_attn(
    const unsigned short* __restrict__ Abf,   // tg_dec bf16 [G,D]
    const unsigned short* __restrict__ Bbf,   // tf_base bf16 [T,D]
    const int* __restrict__ mm,               // motif_mask [G,T]
    const float* __restrict__ tfT,            // tf_expr^T [T,B]
    const float* __restrict__ scaleP,         // [1]
    float* __restrict__ out) {
  extern __shared__ float smemF[];
  float* simLDS = smemF;                                   // [16][kSimStride]
  int*   selI   = (int*)(smemF + kMT * kSimStride);        // [16][32]
  float* selW   = (float*)(selI + kMT * 32);               // [16][32]

  const int tid  = threadIdx.x;
  const int wave = tid >> 5;
  const int lane = tid & 31;
  const int r16  = lane & 15;     // row-within-tile for A / col for B
  const int kh   = lane >> 4;     // half-wave K split
  const int g0   = blockIdx.x * kMT;
  const float scale0 = scaleP[0];

  // ======== Phase 1: sim tile GEMM via v_wmma_f32_16x16x32_bf16 ============
  const unsigned short* Arow = Abf + (size_t)(g0 + r16) * kD;
  for (int it = 0; it < kTilesPerWave; ++it) {
    const int t0 = (wave + kWaves * it) * 16;
    const unsigned short* Brow = Bbf + (size_t)(t0 + r16) * kD;
    if (it + 1 < kTilesPerWave) {
      // hint next B tile into cache (global_prefetch_b8)
      __builtin_prefetch(Bbf + (size_t)(t0 + kWaves * 16 + r16) * kD, 0, 3);
    }
    v8f acc = {0.f, 0.f, 0.f, 0.f, 0.f, 0.f, 0.f, 0.f};
#pragma unroll
    for (int kk = 0; kk < kD / 32; ++kk) {
      const int k0 = kk * 32 + kh * 8;   // elems 0-7 at K=k0..k0+7
      Frag a, b;
      a.q[0] = *(const u32x4*)(Arow + k0);
      a.q[1] = *(const u32x4*)(Arow + k0 + 16);  // elems 8-15 at K+16
      b.q[0] = *(const u32x4*)(Brow + k0);
      b.q[1] = *(const u32x4*)(Brow + k0 + 16);
      acc = __builtin_amdgcn_wmma_f32_16x16x32_bf16(
          false, a.v, false, b.v, (short)0, acc, false, false);
    }
    // C layout: VGPR v -> M = v + 8*kh, N = lane%16
    const int t = t0 + r16;
#pragma unroll
    for (int v = 0; v < 8; ++v) {
      const int m = v + 8 * kh;
      const float s = acc[v] * kRsqrtD;
      const int msk = mm[(size_t)(g0 + m) * kT + t];
      simLDS[m * kSimStride + t] = msk ? s : NEG_INF;
    }
  }
  __syncthreads();

  // ======== Phase 2: per-row softmax + exact top-32 + renorm + store =======
  for (int rr = 0; rr < 2; ++rr) {
    const int row = 2 * wave + rr;
    float* srow = simLDS + row * kSimStride;

    // row max (for exp stability)
    float rmax = NEG_INF;
    for (int j = 0; j < kT / 32; ++j) rmax = fmaxf(rmax, srow[lane + 32 * j]);
#pragma unroll
    for (int off = 16; off > 0; off >>= 1)
      rmax = fmaxf(rmax, __shfl_xor(rmax, off, 32));

    if (rmax == NEG_INF) {
      // fully masked row -> attn row is all zeros (nan_to_num path)
      selI[row * 32 + lane] = 0;
      selW[row * 32 + lane] = 0.f;
    } else {
      // per-lane best candidate over its 48 strided slots
      float bv = NEG_INF; int bp = lane;
      for (int j = 0; j < kT / 32; ++j) {
        const int t = lane + 32 * j;
        const float v2 = srow[t];
        if (v2 > bv) { bv = v2; bp = t; }
      }
      // 32 rounds of wave-argmax (tie -> smaller index, matches stable top_k)
      for (int k = 0; k < kTopK; ++k) {
        float cv = bv; int cp = bp;
#pragma unroll
        for (int off = 16; off > 0; off >>= 1) {
          const float ov = __shfl_xor(cv, off, 32);
          const int   op = __shfl_xor(cp, off, 32);
          if (ov > cv || (ov == cv && op < cp)) { cv = ov; cp = op; }
        }
        if (lane == 0) {
          selI[row * 32 + k] = cp;
          selW[row * 32 + k] = __expf(cv - rmax);  // un-normalized weight
        }
        if ((cp & 31) == lane) {   // owning lane removes winner + rescans
          srow[cp] = NEG_INF;
          bv = NEG_INF; bp = lane;
          for (int j = 0; j < kT / 32; ++j) {
            const int t = lane + 32 * j;
            const float v2 = srow[t];
            if (v2 > bv) { bv = v2; bp = t; }
          }
        }
      }
    }

    // zero the row, normalize over the selected set only (softmax S cancels),
    // scatter 32 weights, stream the row out coalesced (float4)
    for (int j = 0; j < kT / 32; ++j) srow[lane + 32 * j] = 0.f;
    float sw = selW[row * 32 + lane];       // lane == k slot (TOPK == 32)
    float dsum = sw;
#pragma unroll
    for (int off = 16; off > 0; off >>= 1) dsum += __shfl_xor(dsum, off, 32);
    const float wn = sw / (dsum + 1e-30f);
    selW[row * 32 + lane] = wn;
    srow[selI[row * 32 + lane]] = wn;

    float* attnRow = out + (size_t)kB * kG + (size_t)(g0 + row) * kT;
#pragma unroll
    for (int jj = 0; jj < kT / 128; ++jj) {
      const int c = lane * 4 + 128 * jj;
      *(f32x4*)(attnRow + c) = *(const f32x4*)(srow + c);
    }
  }
  __syncthreads();

  // ======== Phase 3: sparse combine tf_scalar[b, g] ========================
  // thread b accumulates 16 outputs from 32 (idx, w) pairs; tfT columns are
  // contiguous so loads are fully coalesced across the block.
  float accv[kMT];
#pragma unroll
  for (int row = 0; row < kMT; ++row) {
    float a = 0.f;
#pragma unroll 4
    for (int j = 0; j < kTopK; ++j) {
      a += selW[row * 32 + j] * tfT[(size_t)selI[row * 32 + j] * kB + tid];
    }
    accv[row] = a;
  }
  // transpose through LDS (reuse sim area) so global stores are coalesced
  float* tile = simLDS;  // [256][17]
#pragma unroll
  for (int row = 0; row < kMT; ++row) tile[tid * 17 + row] = accv[row] * scale0;
  __syncthreads();
  const int lo = tid & 15, hi = tid >> 4;
#pragma unroll
  for (int p = 0; p < 16; ++p) {
    const int b = hi + 16 * p;
    out[(size_t)b * kG + g0 + lo] = tile[b * 17 + lo];
  }
}

// ---------------------------------------------------------------------------
extern "C" void kernel_launch(void* const* d_in, const int* in_sizes, int n_in,
                              void* d_out, int out_size, void* d_ws, size_t ws_size,
                              hipStream_t stream) {
  (void)in_sizes; (void)n_in; (void)out_size; (void)ws_size;
  const float* tg = (const float*)d_in[0];   // tg_dec   [G,D]
  const float* tb = (const float*)d_in[1];   // tf_base  [T,D]
  const float* te = (const float*)d_in[2];   // tf_expr  [B,T]
  const int*   mm = (const int*)d_in[3];     // motif_mask [G,T]
  const float* sc = (const float*)d_in[4];   // scale [1]
  float* out = (float*)d_out;

  char* ws = (char*)d_ws;
  unsigned short* Abf = (unsigned short*)ws;                              // 20.48 MB
  unsigned short* Bbf = (unsigned short*)(ws + (size_t)kG * kD * 2);      // 1.57 MB
  float* tfT = (float*)(ws + (size_t)kG * kD * 2 + (size_t)kT * kD * 2);  // 1.57 MB

  cvt_f32_to_bf16<<<(kG * kD + 255) / 256, 256, 0, stream>>>(tg, Abf, kG * kD);
  cvt_f32_to_bf16<<<(kT * kD + 255) / 256, 256, 0, stream>>>(tb, Bbf, kT * kD);
  transpose_bt<<<(kB * kT + 255) / 256, 256, 0, stream>>>(te, tfT);

  const size_t smem = (size_t)kMT * kSimStride * sizeof(float)  // sim: ~98.6 KB
                    + (size_t)kMT * 32 * (sizeof(int) + sizeof(float));
  fused_attn<<<kG / kMT, 256, smem, stream>>>(Abf, Bbf, mm, tfT, sc, out);
}
